// LFM2RayMLP_73615739453557
// MI455X (gfx1250) — matrix operand order repaired and checked by reading, not compiled, not run on hardware
//
#include <hip/hip_runtime.h>

// Monarch (block-diagonal) SwiGLU MLP for MI455X (gfx1250), bf16 WMMA path.
// B=4, S=4096, D=2048, NB=4, FF=8192 -> T=16384 tokens,
// per-block: in_blk P=512, ff_blk Q=2048, out_blk P=512.

typedef __attribute__((ext_vector_type(16))) __bf16        v16bf;
typedef __attribute__((ext_vector_type(2)))  __bf16        v2bf;
typedef __attribute__((ext_vector_type(8)))  float         v8f;
typedef __attribute__((ext_vector_type(4)))  unsigned int  v4u;

struct Frag32B { v4u lo, hi; };  // 32 bytes == v16bf

static __device__ __forceinline__ unsigned short f32_to_bf16_rne(float f) {
  unsigned int u = __builtin_bit_cast(unsigned int, f);
  unsigned int r = u + 0x7FFFu + ((u >> 16) & 1u);  // round-to-nearest-even
  return (unsigned short)(r >> 16);
}

static __device__ __forceinline__ unsigned int pack2_bf16(float a, float b) {
#if __has_builtin(__builtin_amdgcn_cvt_pk_bf16_f32)
  v2bf p = __builtin_amdgcn_cvt_pk_bf16_f32(a, b);   // v_cvt_pk_bf16_f32
  return __builtin_bit_cast(unsigned int, p);
#else
  return (unsigned int)f32_to_bf16_rne(a) |
         ((unsigned int)f32_to_bf16_rne(b) << 16);
#endif
}

static __device__ __forceinline__ v16bf load_frag(const unsigned short* p0,
                                                  const unsigned short* p1) {
  Frag32B f;
  f.lo = *(const v4u*)p0;   // ds_load_b128
  f.hi = *(const v4u*)p1;   // ds_load_b128
  return __builtin_bit_cast(v16bf, f);
}

static __device__ __forceinline__ v8f wmma_bf16(v16bf a, v16bf b, v8f c) {
  return __builtin_amdgcn_wmma_f32_16x16x32_bf16(
      /*neg_a=*/false, a, /*neg_b=*/false, b,
      /*c_mod=*/(short)0, c, /*reuse_a=*/false, /*reuse_b=*/false);
}

// Pin the K-step schedule: NLOADS LDS reads first, then NWMMA back-to-back WMMAs.
// (builtin requires integer-constant args -> template parameters)
template <int NLOADS, int NWMMA>
static __device__ __forceinline__ void sched_ds_then_wmma() {
#if __has_builtin(__builtin_amdgcn_sched_group_barrier)
  __builtin_amdgcn_sched_group_barrier(0x100, NLOADS, 0);  // DS read group
  __builtin_amdgcn_sched_group_barrier(0x008, NWMMA, 0);   // MFMA/WMMA group
#endif
}

// CDNA5 async global->LDS copy, 16 bytes per lane (tracked by ASYNCcnt).
static __device__ __forceinline__ void async_copy_b128(unsigned int lds_byte_addr,
                                                       const void* gaddr) {
  asm volatile("global_load_async_to_lds_b128 %0, %1, off"
               :: "v"(lds_byte_addr), "v"(gaddr)
               : "memory");
}

static __device__ __forceinline__ void wait_asynccnt0() {
  asm volatile("s_wait_asynccnt 0x0" ::: "memory");
}

// ---------------------------------------------------------------------------
// Stage 1: per block n, gate = X_n @ W1_n^T, up = X_n @ W3_n^T (shared A tile),
//          H_n = silu(gate) * up  -> bf16 in workspace, layout (T, NB*Q).
// Workgroup tile: M=128 (8 waves x 16 rows), N=64 (4 wmma col-tiles/wave), K step 32.
// ---------------------------------------------------------------------------
__global__ __launch_bounds__(256)
void monarch_swiglu_stage1(const float* __restrict__ x,
                           const float* __restrict__ w1,
                           const float* __restrict__ w3,
                           unsigned short* __restrict__ h) {
  constexpr int D = 2048, P = 512, Q = 2048, FF = 8192;
  constexpr int MT = 128, NT = 64, KT = 32;

  __shared__ __align__(16) unsigned short sA [MT * KT];  // [m][k] bf16, 8KB
  __shared__ __align__(16) unsigned short sB1[NT * KT];  // [n][k] bf16, 4KB
  __shared__ __align__(16) unsigned short sB3[NT * KT];  // 4KB

  const int tid  = threadIdx.x;
  const int wave = tid >> 5;
  const int lane = tid & 31;
  const int t0   = blockIdx.x * MT;   // token tile
  const int n0   = blockIdx.y * NT;   // ff-column tile within block
  const int blk  = blockIdx.z;        // diagonal block

  const float* xb  = x  + (size_t)t0 * D + (size_t)blk * P;   // row stride D
  const float* w1b = w1 + ((size_t)blk * Q + n0) * P;         // row stride P
  const float* w3b = w3 + ((size_t)blk * Q + n0) * P;

  v8f accg[4] = {};
  v8f accu[4] = {};

  // A-frag addressing (16-bit A 16x32 layout): lane half selects K-phase.
  const int am = wave * 16 + (lane & 15);
  const int kh = (lane >> 4) * 8;           // K offset 0 or 8
  // B-frag addressing (16-bit B 32x16 layout): lane = column, same K-phase as A.
  const int bn = lane & 15;
  const int bk = (lane >> 4) * 8;           // K offset 0 or 8

  for (int k0 = 0; k0 < P; k0 += KT) {
    __syncthreads();
    // Stage A tile: 128x32 f32 -> bf16 packed pairs (8 dword stores/thread)
    for (int i = tid; i < MT * (KT / 2); i += 256) {
      const int m = i >> 4, kp = i & 15;
      const float2 v = *(const float2*)&xb[(size_t)m * D + k0 + kp * 2];
      ((unsigned int*)sA)[m * (KT / 2) + kp] = pack2_bf16(v.x, v.y);
      if (k0 + KT < P)  // global_prefetch next K-tile of activations
        __builtin_prefetch(&xb[(size_t)m * D + k0 + KT + kp * 2], 0, 0);
    }
    // Stage W1/W3 tiles: 64x32 each (4 dword stores/thread each)
    for (int i = tid; i < NT * (KT / 2); i += 256) {
      const int n = i >> 4, kp = i & 15;
      const float2 a = *(const float2*)&w1b[(size_t)n * P + k0 + kp * 2];
      ((unsigned int*)sB1)[n * (KT / 2) + kp] = pack2_bf16(a.x, a.y);
      const float2 b = *(const float2*)&w3b[(size_t)n * P + k0 + kp * 2];
      ((unsigned int*)sB3)[n * (KT / 2) + kp] = pack2_bf16(b.x, b.y);
    }
    __syncthreads();

    // Load ALL fragments (18 ds_load_b128), then fire 8 back-to-back WMMAs.
    const v16bf afrag = load_frag(&sA[am * KT + kh], &sA[am * KT + 16 + kh]);
    v16bf bfrag1[4], bfrag3[4];
#pragma unroll
    for (int j = 0; j < 4; ++j) {
      const unsigned short* b1p = &sB1[(j * 16 + bn) * KT + bk];
      bfrag1[j] = load_frag(b1p, b1p + 16);
      const unsigned short* b3p = &sB3[(j * 16 + bn) * KT + bk];
      bfrag3[j] = load_frag(b3p, b3p + 16);
    }
#pragma unroll
    for (int j = 0; j < 4; ++j) {
      accg[j] = wmma_bf16(afrag, bfrag1[j], accg[j]);
      accu[j] = wmma_bf16(afrag, bfrag3[j], accu[j]);
    }
    sched_ds_then_wmma<18, 8>();
  }

  // Epilogue: H = silu(gate) * up, bf16 store.
  // C/D layout: VGPR r holds row m = r + 8*(lane>>4), col = lane&15.
  const int mrow = t0 + wave * 16 + (lane >> 4) * 8;
  const int ccol = blk * Q + n0 + (lane & 15);
#pragma unroll
  for (int j = 0; j < 4; ++j) {
#pragma unroll
    for (int r = 0; r < 8; ++r) {
      const float g = accg[j][r];
      const float u = accu[j][r];
      // silu(g) = g * sigmoid(g); v_rcp_f32 is plenty of precision here
      const float s = g * __builtin_amdgcn_rcpf(1.0f + __expf(-g));
      h[(size_t)(mrow + r) * FF + (ccol + j * 16)] = f32_to_bf16_rne(s * u);
    }
  }
}

// ---------------------------------------------------------------------------
// Stage 2: per block n, Out_n = H_n (T x 2048, bf16) @ W2_n^T (512 x 2048 f32).
// A tile is already bf16 -> moved with GLOBAL_LOAD_ASYNC_TO_LDS_B128 (ASYNCcnt).
// Same tiling: M=128, N=64, K step 32, 64 K iterations.
// ---------------------------------------------------------------------------
__global__ __launch_bounds__(256)
void monarch_swiglu_stage2(const unsigned short* __restrict__ h,
                           const float* __restrict__ w2,
                           float* __restrict__ out) {
  constexpr int D = 2048, P = 512, Q = 2048, FF = 8192;
  constexpr int MT = 128, NT = 64, KT = 32;

  __shared__ __align__(16) unsigned short sA[MT * KT];  // 8KB
  __shared__ __align__(16) unsigned short sB[NT * KT];  // 4KB

  const int tid  = threadIdx.x;
  const int wave = tid >> 5;
  const int lane = tid & 31;
  const int t0   = blockIdx.x * MT;
  const int n0   = blockIdx.y * NT;   // output column tile within block (0..511)
  const int blk  = blockIdx.z;

  const unsigned short* hb = h + (size_t)t0 * FF + (size_t)blk * Q;  // row stride FF
  const float* w2b = w2 + ((size_t)blk * P + n0) * Q;                // row stride Q

  const unsigned int sA_base = (unsigned int)(uintptr_t)&sA[0];

  v8f acc[4] = {};

  const int am = wave * 16 + (lane & 15);
  const int kh = (lane >> 4) * 8;
  const int bn = lane & 15;
  const int bk = (lane >> 4) * 8;

  for (int k0 = 0; k0 < Q; k0 += KT) {
    __syncthreads();
    // A tile (bf16 H): 512 x 16B chunks, async DMA to LDS (2 per thread).
    for (int i = tid; i < MT * KT / 8; i += 256) {
      const int m = i >> 2, q = i & 3;            // row, 16B chunk within row
      async_copy_b128(sA_base + (unsigned int)((m * KT + q * 8) * 2),
                      &hb[(size_t)m * FF + k0 + q * 8]);
    }
    // W2 tile: f32 -> bf16 (4 dword stores/thread)
    for (int i = tid; i < NT * (KT / 2); i += 256) {
      const int n = i >> 4, kp = i & 15;
      const float2 v = *(const float2*)&w2b[(size_t)n * Q + k0 + kp * 2];
      ((unsigned int*)sB)[n * (KT / 2) + kp] = pack2_bf16(v.x, v.y);
    }
    wait_asynccnt0();   // our async LDS writes complete
    __syncthreads();    // visible workgroup-wide

    const v16bf afrag = load_frag(&sA[am * KT + kh], &sA[am * KT + 16 + kh]);
    v16bf bfrag[4];
#pragma unroll
    for (int j = 0; j < 4; ++j) {
      const unsigned short* bp = &sB[(j * 16 + bn) * KT + bk];
      bfrag[j] = load_frag(bp, bp + 16);
    }
#pragma unroll
    for (int j = 0; j < 4; ++j) {
      acc[j] = wmma_bf16(afrag, bfrag[j], acc[j]);
    }
    sched_ds_then_wmma<10, 4>();
  }

  const int mrow = t0 + wave * 16 + (lane >> 4) * 8;
  const int ccol = blk * P + n0 + (lane & 15);
#pragma unroll
  for (int j = 0; j < 4; ++j) {
#pragma unroll
    for (int r = 0; r < 8; ++r) {
      out[(size_t)(mrow + r) * D + (ccol + j * 16)] = acc[j][r];
    }
  }
}

extern "C" void kernel_launch(void* const* d_in, const int* in_sizes, int n_in,
                              void* d_out, int out_size, void* d_ws, size_t ws_size,
                              hipStream_t stream) {
  const float* x  = (const float*)d_in[0];
  const float* w1 = (const float*)d_in[1];
  const float* w3 = (const float*)d_in[2];
  const float* w2 = (const float*)d_in[3];
  float* out = (float*)d_out;

  const int D = 2048, FF = 8192, NB = 4;
  const int T = in_sizes[0] / D;            // 16384 tokens

  unsigned short* h = (unsigned short*)d_ws;  // T x FF bf16 intermediate (256 MB)

  dim3 block(256);
  dim3 g1(T / 128, (FF / NB) / 64, NB);     // (128, 32, 4)
  monarch_swiglu_stage1<<<g1, block, 0, stream>>>(x, w1, w3, h);

  dim3 g2(T / 128, (D / NB) / 64, NB);      // (128, 8, 4)
  monarch_swiglu_stage2<<<g2, block, 0, stream>>>(h, w2, out);
}
